// ConvHex_2121713844833
// MI455X (gfx1250) — compile-verified, matrix operand order checked by reading, not run on hardware
//
#include <hip/hip_runtime.h>
#include <hip/hip_bf16.h>
#include <stdint.h>
#include <stddef.h>

#define B_    256
#define CIN   64
#define COUT  128
#define H_    1039
#define HP    1040      // padded: row H_ is all-zero, used for -1 neighbors
#define KNB   6
#define SLOTS 7         // center + 6 neighbors
#define KD    448       // SLOTS * CIN

#define PANEL_COLS   32                       // batch columns per block
#define COL_BYTES    (KD * 2)                 // 896 B per column (bf16)
#define PANEL_BYTES  (PANEL_COLS * COL_BYTES) // 28672 B (hi); lo mirrored after
#define LDS_BYTES    (2 * PANEL_BYTES)        // 57344 B

typedef __attribute__((ext_vector_type(16))) __bf16       v16bf;
typedef __attribute__((ext_vector_type(8)))  float        v8f;
typedef __attribute__((ext_vector_type(4)))  unsigned int u32x4;
typedef __attribute__((ext_vector_type(4)))  int          i32x4;

typedef __attribute__((address_space(1))) i32x4* as1_i32x4_p;
typedef __attribute__((address_space(3))) i32x4* as3_i32x4_p;

union Frag { v16bf v; u32x4 u[2]; };

// ---- CDNA5 async copy global->LDS (ASYNCcnt-tracked) ----------------------
__device__ __forceinline__ void async_b128(const void* gptr, const void* lptr) {
#if __has_builtin(__builtin_amdgcn_global_load_async_to_lds_b128)
  __builtin_amdgcn_global_load_async_to_lds_b128(
      (as1_i32x4_p)(uintptr_t)gptr,
      (as3_i32x4_p)(unsigned)(uintptr_t)lptr, 0, 0);
#else
  asm volatile("global_load_async_to_lds_b128 %0, %1, off"
               :: "v"((unsigned)(uintptr_t)lptr),
                  "v"((unsigned long long)(uintptr_t)gptr)
               : "memory");
#endif
}

__device__ __forceinline__ void wait_async0() {
#if __has_builtin(__builtin_amdgcn_s_wait_asynccnt)
  __builtin_amdgcn_s_wait_asynccnt(0);
#else
  asm volatile("s_wait_asynccnt 0" ::: "memory");
#endif
}

// ---------------------------------------------------------------------------
// Kernel 1: LDS-tiled transpose [B, CIN, H] f32 -> [B, HP, CIN] bf16 hi/lo
// (pad row h==H_ is zero so that neighbor index -1 gathers zeros)
// ---------------------------------------------------------------------------
__global__ void prep_x(const float* __restrict__ x,
                       __hip_bfloat16* __restrict__ xhi,
                       __hip_bfloat16* __restrict__ xlo) {
  __shared__ float tile[32][33];
  const int b  = blockIdx.z;
  const int h0 = blockIdx.x * 32;
  const int c0 = blockIdx.y * 32;
  const int tx = threadIdx.x;   // 0..31
  const int ty = threadIdx.y;   // 0..7

#pragma unroll
  for (int i = ty; i < 32; i += 8) {
    const int c = c0 + i;
    const int h = h0 + tx;
    tile[i][tx] = (h < H_) ? x[((size_t)b * CIN + c) * H_ + h] : 0.0f;
  }
  __syncthreads();

#pragma unroll
  for (int i = ty; i < 32; i += 8) {
    const int h = h0 + i;
    const int c = c0 + tx;
    if (h < HP) {
      const float v = tile[tx][i];
      const __hip_bfloat16 hi = __float2bfloat16(v);
      const size_t idx = ((size_t)b * HP + h) * CIN + c;
      xhi[idx] = hi;
      xlo[idx] = __float2bfloat16(v - __bfloat162float(hi));
    }
  }
}

// ---------------------------------------------------------------------------
// Kernel 2: pack combined weights W[o, slot*64 + c] as bf16 hi/lo
// ---------------------------------------------------------------------------
__global__ void prep_w(const float* __restrict__ wc,
                       const float* __restrict__ wn,
                       __hip_bfloat16* __restrict__ whi,
                       __hip_bfloat16* __restrict__ wlo) {
  const int t = blockIdx.x * blockDim.x + threadIdx.x;
  if (t >= COUT * KD) return;
  const int k    = t % KD;
  const int o    = t / KD;
  const int slot = k / CIN;
  const int c    = k % CIN;
  const float v = (slot == 0) ? wc[o * CIN + c]
                              : wn[(o * CIN + c) * KNB + (slot - 1)];
  const __hip_bfloat16 hi = __float2bfloat16(v);
  whi[t] = hi;
  wlo[t] = __float2bfloat16(v - __bfloat162float(hi));
}

// ---------------------------------------------------------------------------
// Kernel 3: WMMA GEMM with async-staged LDS B panel.
// Block = 8 waves (256 thr), one h, 32 batch columns.
// Stage once: 57 KB gathered panel via global_load_async_to_lds_b128
//             (7 slots -> slot index is loop-constant, rows[] stays in regs).
// Then 14 K-steps x 6 v_wmma_f32_16x16x32_bf16 per wave, fragments via
// ds_load_b128 (B) and global_load_b128 (A).
// ---------------------------------------------------------------------------
__global__ void __launch_bounds__(256) hexconv_gemm(
    const __hip_bfloat16* __restrict__ xhi,
    const __hip_bfloat16* __restrict__ xlo,
    const __hip_bfloat16* __restrict__ whi,
    const __hip_bfloat16* __restrict__ wlo,
    const int* __restrict__ neighbors,
    const float* __restrict__ bias,
    float* __restrict__ out) {
  __shared__ char lds_panel[LDS_BYTES] __attribute__((aligned(64)));

  const int h    = blockIdx.y;
  const int b0   = blockIdx.x * PANEL_COLS;
  const int wave = threadIdx.x >> 5;
  const int lane = threadIdx.x & 31;
  const int half = lane >> 4;       // K-half selector per ISA 16-bit layout
  const int lid  = lane & 15;       // M row (A) / N col (B,D)
  const int m0   = wave * 16;

  // gather rows for the 7 slots; -1 -> zero pad row H_
  int rows[SLOTS];
  rows[0] = h;
  int cnt = 1;
#pragma unroll
  for (int k = 0; k < KNB; ++k) {
    const int nb = neighbors[h * KNB + k];
    rows[k + 1] = (nb >= 0) ? nb : H_;
    cnt += (nb >= 0) ? 1 : 0;
  }
  const float scale = 1.0f / (float)cnt;

  const size_t colstride = (size_t)HP * CIN;

  // ---- stage B panel: per slot j, 256 threads cover 32 cols x 64 ch ------
  {
    const int col = threadIdx.x >> 3;        // 0..31
    const int c8  = (threadIdx.x & 7) * 8;   // element offset within slot
    const size_t gcol = (size_t)(b0 + col) * colstride;
    const unsigned lcol = (unsigned)col * COL_BYTES;
#pragma unroll
    for (int j = 0; j < SLOTS; ++j) {        // slot index compile-time const
      const size_t goff = gcol + (size_t)rows[j] * CIN + c8;
      const unsigned loff = lcol + (unsigned)(j * CIN + c8) * 2;
      async_b128(xhi + goff, lds_panel + loff);
      async_b128(xlo + goff, lds_panel + PANEL_BYTES + loff);
    }
  }
  wait_async0();
  __syncthreads();

  v8f acc0 = {};   // N tile 0 (b0 .. b0+15)
  v8f acc1 = {};   // N tile 1 (b0+16 .. b0+31)

  const __hip_bfloat16* wh = whi + (size_t)(m0 + lid) * KD;
  const __hip_bfloat16* wl = wlo + (size_t)(m0 + lid) * KD;

#pragma unroll
  for (int kt = 0; kt < KD / 32; ++kt) {
    const int k0 = kt * 32;

    // A fragments: per-lane K = k0 + 8*half + [0..7], then +16 more
    Frag ah, al;
    ah.u[0] = *(const u32x4*)(wh + k0 + 8 * half);
    ah.u[1] = *(const u32x4*)(wh + k0 + 16 + 8 * half);
    al.u[0] = *(const u32x4*)(wl + k0 + 8 * half);
    al.u[1] = *(const u32x4*)(wl + k0 + 16 + 8 * half);

    // B fragments from LDS panel (col-major rows of 896 B)
    const char* pb0 = lds_panel + (unsigned)lid        * COL_BYTES + (k0 + 8 * half) * 2;
    const char* pb1 = lds_panel + (unsigned)(16 + lid) * COL_BYTES + (k0 + 8 * half) * 2;
    Frag bh0, bl0, bh1, bl1;
    bh0.u[0] = *(const u32x4*)(pb0);
    bh0.u[1] = *(const u32x4*)(pb0 + 32);
    bl0.u[0] = *(const u32x4*)(pb0 + PANEL_BYTES);
    bl0.u[1] = *(const u32x4*)(pb0 + PANEL_BYTES + 32);
    bh1.u[0] = *(const u32x4*)(pb1);
    bh1.u[1] = *(const u32x4*)(pb1 + 32);
    bl1.u[0] = *(const u32x4*)(pb1 + PANEL_BYTES);
    bl1.u[1] = *(const u32x4*)(pb1 + PANEL_BYTES + 32);

    // split-precision product: hi*hi + hi*lo + lo*hi (f32 accumulate)
    acc0 = __builtin_amdgcn_wmma_f32_16x16x32_bf16(false, ah.v, false, bh0.v, (short)0, acc0, false, false);
    acc0 = __builtin_amdgcn_wmma_f32_16x16x32_bf16(false, ah.v, false, bl0.v, (short)0, acc0, false, false);
    acc0 = __builtin_amdgcn_wmma_f32_16x16x32_bf16(false, al.v, false, bh0.v, (short)0, acc0, false, false);
    acc1 = __builtin_amdgcn_wmma_f32_16x16x32_bf16(false, ah.v, false, bh1.v, (short)0, acc1, false, false);
    acc1 = __builtin_amdgcn_wmma_f32_16x16x32_bf16(false, ah.v, false, bl1.v, (short)0, acc1, false, false);
    acc1 = __builtin_amdgcn_wmma_f32_16x16x32_bf16(false, al.v, false, bh1.v, (short)0, acc1, false, false);
  }

  // Epilogue: D layout VGPR r <-> M = r + 8*half, lane <-> N
#pragma unroll
  for (int r = 0; r < 8; ++r) {
    const int m  = m0 + r + 8 * half;
    const float bv = bias[m];
    const int bb = b0 + lid;
    out[((size_t)bb * COUT + m) * H_ + h]        = acc0[r] * scale + bv;
    out[((size_t)(bb + 16) * COUT + m) * H_ + h] = acc1[r] * scale + bv;
  }
}

// ---------------------------------------------------------------------------
extern "C" void kernel_launch(void* const* d_in, const int* in_sizes, int n_in,
                              void* d_out, int out_size, void* d_ws, size_t ws_size,
                              hipStream_t stream) {
  const float* x         = (const float*)d_in[0];
  const int*   neighbors = (const int*)d_in[1];
  const float* wc        = (const float*)d_in[2];
  const float* wn        = (const float*)d_in[3];
  const float* bias      = (const float*)d_in[4];
  float*       out       = (float*)d_out;

  // workspace layout (bf16): xhi, xlo [B*HP*CIN], whi, wlo [COUT*KD]
  const size_t XT = (size_t)B_ * HP * CIN;
  __hip_bfloat16* xhi = (__hip_bfloat16*)d_ws;
  __hip_bfloat16* xlo = xhi + XT;
  __hip_bfloat16* whi = xlo + XT;
  __hip_bfloat16* wlo = whi + (size_t)COUT * KD;

  {
    dim3 grid((H_ + 31) / 32, CIN / 32, B_);   // 33 x 2 x 256
    dim3 block(32, 8);
    prep_x<<<grid, block, 0, stream>>>(x, xhi, xlo);
  }
  {
    const int total = COUT * KD;
    prep_w<<<(total + 255) / 256, 256, 0, stream>>>(wc, wn, whi, wlo);
  }
  {
    dim3 grid(B_ / PANEL_COLS, H_);            // 8 x 1039 workgroups
    hexconv_gemm<<<grid, 256, 0, stream>>>(xhi, xlo, whi, wlo, neighbors, bias, out);
  }
}
